// LMAB_2302102471288
// MI455X (gfx1250) — matrix operand order repaired and checked by reading, not compiled, not run on hardware
//
#include <hip/hip_runtime.h>
#include <hip/hip_bf16.h>
#include <math.h>

typedef _Float16 half_t;
typedef __attribute__((ext_vector_type(16))) _Float16 v16h;
typedef __attribute__((ext_vector_type(8)))  _Float16 v8h;
typedef __attribute__((ext_vector_type(8)))  float    v8f;

// Problem constants
constexpr int BB = 16, CC = 256, HH = 40, WWD = 40;
constexpr int NPIX = BB * HH * WWD;      // 25600
constexpr int HID = 512;

// ---------------------------------------------------------------------------
// Layout transforms
// ---------------------------------------------------------------------------
__global__ void k_nchw_to_pix(const float* __restrict__ x, float* __restrict__ out) {
  int i = blockIdx.x * 256 + threadIdx.x;             // input (NCHW) index
  if (i >= BB * CC * HH * WWD) return;
  int hw = i % (HH * WWD);
  int t  = i / (HH * WWD);
  int c  = t % CC;
  int b  = t / CC;
  out[((size_t)(b * HH * WWD + hw)) * CC + c] = x[i];
}

__global__ void k_pix_to_nchw(const float* __restrict__ xpix, float* __restrict__ out) {
  int i = blockIdx.x * 256 + threadIdx.x;             // output (NCHW) index
  if (i >= BB * CC * HH * WWD) return;
  int hw = i % (HH * WWD);
  int t  = i / (HH * WWD);
  int c  = t % CC;
  int b  = t / CC;
  out[i] = xpix[((size_t)(b * HH * WWD + hw)) * CC + c];
}

__global__ void k_cast_f16(const float* __restrict__ src, half_t* __restrict__ dst, int n) {
  int i = blockIdx.x * 256 + threadIdx.x;
  if (i < n) dst[i] = (half_t)src[i];
}

// ---------------------------------------------------------------------------
// LayerNorm2d over channels (pixel-major).  One wave per pixel, 8 ch/lane.
// ---------------------------------------------------------------------------
template <bool OUT16>
__global__ void k_ln(const float* __restrict__ x, const float* __restrict__ w,
                     const float* __restrict__ bias, half_t* __restrict__ o16,
                     float* __restrict__ o32) {
  int wave = blockIdx.x * 8 + (threadIdx.x >> 5);
  int lane = threadIdx.x & 31;
  if (wave >= NPIX) return;
  const float* row = x + (size_t)wave * CC;
  float v[8];
  float s = 0.f, s2 = 0.f;
#pragma unroll
  for (int i = 0; i < 8; i++) {
    float t = row[lane + 32 * i];
    v[i] = t; s += t; s2 += t * t;
  }
#pragma unroll
  for (int off = 16; off; off >>= 1) {
    s  += __shfl_xor(s,  off, 32);
    s2 += __shfl_xor(s2, off, 32);
  }
  float mu  = s * (1.0f / CC);
  float var = s2 * (1.0f / CC) - mu * mu;
  float rs  = rsqrtf(var + 1e-6f);
#pragma unroll
  for (int i = 0; i < 8; i++) {
    int c = lane + 32 * i;
    float y = (v[i] - mu) * rs * w[c] + bias[c];
    if (OUT16) o16[(size_t)wave * CC + c] = (half_t)y;
    else       o32[(size_t)wave * CC + c] = y;
  }
}

// ---------------------------------------------------------------------------
// WMMA GEMM: out[n,o] = sum_k A[n,k] * W[o,k] + bias[o] (+ res[n,o])
// A: (NPIX x 256) f16 row-major.  W: (O x 256) f16 row-major (acts as B^T).
// Wave computes 16 (M) x 64 (O).  K fixed at 256.
// MODE 0: f32 out + bias.  MODE 1: f32 out + bias + residual.  MODE 2: f16 out + bias.
// ---------------------------------------------------------------------------
template <int MODE>
__global__ void k_gemm(const half_t* __restrict__ A, const half_t* __restrict__ Wt,
                       const float* __restrict__ bias, const float* __restrict__ res,
                       float* __restrict__ out32, half_t* __restrict__ out16, int O) {
  constexpr int K = 256;
  int oTiles = O >> 6;
  int wt = blockIdx.x * 8 + (threadIdx.x >> 5);
  if (wt >= (NPIX / 16) * oTiles) return;
  int lane = threadIdx.x & 31;
  int mt = wt / oTiles, ot = wt % oTiles;
  int lmod = lane & 15, lhi = lane >> 4;

  const half_t* Arow = A + ((size_t)(mt * 16 + lmod)) * K;
  const half_t* Wb[4];
#pragma unroll
  for (int j = 0; j < 4; j++)
    Wb[j] = Wt + ((size_t)(ot * 64 + j * 16 + lmod)) * K + 16 * lhi;

  v8f acc[4];
#pragma unroll
  for (int j = 0; j < 4; j++)
#pragma unroll
    for (int r = 0; r < 8; r++) acc[j][r] = 0.f;

  for (int k0 = 0; k0 < K; k0 += 32) {
    // A fragment (16x32 f16): lane m = lmod; elements e:
    //  e=0..7  -> K = k0 + e + 16*lhi       (contiguous 8 halfs)
    //  e=8..15 -> K = k0 + e + 16*lhi       (contiguous 8 halfs)
    v8h a0 = *(const v8h*)(Arow + k0 + 16 * lhi);
    v8h a1 = *(const v8h*)(Arow + k0 + 16 * lhi + 8);
    v16h a;
#pragma unroll
    for (int i = 0; i < 8; i++) { a[i] = a0[i]; a[8 + i] = a1[i]; }
#pragma unroll
    for (int j = 0; j < 4; j++) {
      // B fragment (32x16 f16): lane n = lmod; element e -> K = k0 + e + 16*lhi
      // => 16 consecutive halfs of W row (K contiguous): single 32B load.
      v16h b = *(const v16h*)(Wb[j] + k0);
      acc[j] = __builtin_amdgcn_wmma_f32_16x16x32_f16(
          false, a, false, b, (short)0, acc[j], false, false);
    }
  }

#pragma unroll
  for (int j = 0; j < 4; j++) {
    int o = ot * 64 + j * 16 + lmod;
    float bv = bias[o];
#pragma unroll
    for (int r = 0; r < 8; r++) {
      int row = mt * 16 + r + 8 * lhi;           // C/D layout: M = r + 8*(lane/16)
      size_t idx = (size_t)row * O + o;
      float val = acc[j][r] + bv;
      if (MODE == 1) val += res[idx];
      if (MODE == 2) out16[idx] = (half_t)val;
      else           out32[idx] = val;
    }
  }
}

// ---------------------------------------------------------------------------
// Local neighborhood attention. One wave per (pixel, head); lane = d (HD==32).
// qkv: (NPIX, 768) f16; q ch 0..255, k ch 256..511, v ch 512..767.
// ---------------------------------------------------------------------------
template <int KSZ, int DIL>
__global__ void k_attn(const half_t* __restrict__ qkv, const float* __restrict__ rpb,
                       half_t* __restrict__ out, int headStart, int nHeads) {
  int wid = blockIdx.x * 8 + (threadIdx.x >> 5);
  if (wid >= NPIX * nHeads) return;
  int lane = threadIdx.x & 31;
  int n = wid / nHeads;
  int hl = wid - n * nHeads;
  int head = headStart + hl;
  int b = n / (HH * WWD);
  int hw = n - b * (HH * WWD);
  int py = hw / WWD, px = hw - py * WWD;

  const float scale = 0.17677669529663687f;       // 32^-0.5
  float q = (float)qkv[(size_t)n * 768 + head * 32 + lane] * scale;

  float logit[KSZ * KSZ];
  constexpr int CTR = KSZ / 2;
#pragma unroll
  for (int i = 0; i < KSZ; i++)
#pragma unroll
    for (int j = 0; j < KSZ; j++) {
      int yy = py + (i - CTR) * DIL, xx = px + (j - CTR) * DIL;
      float kd = 0.f;
      if (yy >= 0 && yy < HH && xx >= 0 && xx < WWD) {
        int nn = b * (HH * WWD) + yy * WWD + xx;
        kd = (float)qkv[(size_t)nn * 768 + 256 + head * 32 + lane];
      }
      float p = q * kd;
#pragma unroll
      for (int off = 16; off; off >>= 1) p += __shfl_xor(p, off, 32);
      logit[i * KSZ + j] = p + rpb[hl * KSZ * KSZ + i * KSZ + j];
    }

  float mx = logit[0];
#pragma unroll
  for (int l = 1; l < KSZ * KSZ; l++) mx = fmaxf(mx, logit[l]);
  float se = 0.f;
#pragma unroll
  for (int l = 0; l < KSZ * KSZ; l++) {
    float e = expf(logit[l] - mx);
    logit[l] = e; se += e;
  }
  float inv = 1.f / se;

  float o = 0.f;
#pragma unroll
  for (int i = 0; i < KSZ; i++)
#pragma unroll
    for (int j = 0; j < KSZ; j++) {
      int yy = py + (i - CTR) * DIL, xx = px + (j - CTR) * DIL;
      if (yy >= 0 && yy < HH && xx >= 0 && xx < WWD) {
        int nn = b * (HH * WWD) + yy * WWD + xx;
        float vd = (float)qkv[(size_t)nn * 768 + 512 + head * 32 + lane];
        o += logit[i * KSZ + j] * vd;
      }
    }
  out[(size_t)n * CC + head * 32 + lane] = (half_t)(o * inv);
}

// ---------------------------------------------------------------------------
// msconvstar depthwise: out = h + dwconv_group(h)  (512 ch, k = 1/3/5/7)
// ---------------------------------------------------------------------------
__global__ void k_mlp_dw(const float* __restrict__ h,
                         const float* __restrict__ w0, const float* __restrict__ w1,
                         const float* __restrict__ w2, const float* __restrict__ w3,
                         const float* __restrict__ b0, const float* __restrict__ b1,
                         const float* __restrict__ b2, const float* __restrict__ b3,
                         float* __restrict__ out) {
  int idx = blockIdx.x * 256 + threadIdx.x;
  if (idx >= NPIX * HID) return;
  int c = idx & (HID - 1);
  int n = idx >> 9;
  int b = n / (HH * WWD);
  int hw = n - b * (HH * WWD);
  int py = hw / WWD, px = hw - py * WWD;
  int g = c >> 7, cl = c & 127;
  const float* wp; const float* bp; int k;
  switch (g) {
    case 0:  wp = w0; bp = b0; k = 1; break;
    case 1:  wp = w1; bp = b1; k = 3; break;
    case 2:  wp = w2; bp = b2; k = 5; break;
    default: wp = w3; bp = b3; k = 7; break;
  }
  int ctr = k >> 1;
  float acc = bp[cl];
  for (int i = 0; i < k; i++)
    for (int j = 0; j < k; j++) {
      int yy = py + i - ctr, xx = px + j - ctr;
      if (yy >= 0 && yy < HH && xx >= 0 && xx < WWD)
        acc += wp[cl * k * k + i * k + j] *
               h[((size_t)(b * (HH * WWD) + yy * WWD + xx)) * HID + c];
    }
  out[idx] = h[idx] + acc;
}

__device__ __forceinline__ float gelu_erf(float x) {
  return 0.5f * x * (1.f + erff(x * 0.70710678118654752f));
}

// g = gelu(h[:,0:256]) * h[:,256:512]  -> f16
__global__ void k_star(const float* __restrict__ hs, half_t* __restrict__ out) {
  int idx = blockIdx.x * 256 + threadIdx.x;
  if (idx >= NPIX * CC) return;
  int c = idx & (CC - 1);
  int n = idx >> 8;
  float a  = hs[(size_t)n * HID + c];
  float b2 = hs[(size_t)n * HID + CC + c];
  out[idx] = (half_t)(gelu_erf(a) * b2);
}

// cascaded_sma stage: out = gelu(dwconv3x3_dil(x) + bias)
__global__ void k_sma_dw(const float* __restrict__ x, const float* __restrict__ w,
                         const float* __restrict__ bias, float* __restrict__ out, int dil) {
  int idx = blockIdx.x * 256 + threadIdx.x;
  if (idx >= NPIX * CC) return;
  int c = idx & (CC - 1);
  int n = idx >> 8;
  int b = n / (HH * WWD);
  int hw = n - b * (HH * WWD);
  int py = hw / WWD, px = hw - py * WWD;
  float acc = bias[c];
#pragma unroll
  for (int i = 0; i < 3; i++)
#pragma unroll
    for (int j = 0; j < 3; j++) {
      int yy = py + (i - 1) * dil, xx = px + (j - 1) * dil;
      if (yy >= 0 && yy < HH && xx >= 0 && xx < WWD)
        acc += w[c * 9 + i * 3 + j] *
               x[((size_t)(b * (HH * WWD) + yy * WWD + xx)) * CC + c];
    }
  out[idx] = gelu_erf(acc);
}

// ---------------------------------------------------------------------------
// Host side: input mapping + orchestration
// ---------------------------------------------------------------------------
enum Name {
  NX = 0, NLN1W, NLN1B, NLN2W, NLN2B, NLN3W, NLN3B, NLN4W, NLN4B,
  NQKVW, NQKVB, NPROJW, NPROJB, NRPB0, NRPB1, NRPB2,
  NM1FC1W, NM1FC1B, NM1DWW0, NM1DWW1, NM1DWW2, NM1DWW3,
  NM1DWB0, NM1DWB1, NM1DWB2, NM1DWB3, NM1FC2W, NM1FC2B,
  NM2FC1W, NM2FC1B, NM2DWW0, NM2DWW1, NM2DWW2, NM2DWW3,
  NM2DWB0, NM2DWB1, NM2DWB2, NM2DWB3, NM2FC2W, NM2FC2B,
  NSMAW0, NSMAW1, NSMAW2, NSMAB0, NSMAB1, NSMAB2, NSMAPWW, NSMAPWB,
  NCOUNT
};

static const int kSizeOf[NCOUNT] = {
  6553600, 256, 256, 256, 256, 256, 256, 256, 256,
  196608, 768, 65536, 256, 45, 50, 49,
  131072, 512, 128, 1152, 3200, 6272, 128, 128, 128, 128, 65536, 256,
  131072, 512, 128, 1152, 3200, 6272, 128, 128, 128, 128, 65536, 256,
  2304, 2304, 2304, 256, 256, 256, 65536, 256
};

// Candidate A: jax.tree_leaves of full dict (all dict keys sorted).
static const int ORD_A[48] = {
  NLN1B, NLN1W, NLN2B, NLN2W, NLN3B, NLN3W, NLN4B, NLN4W,
  NM1DWB0, NM1DWB1, NM1DWB2, NM1DWB3, NM1DWW0, NM1DWW1, NM1DWW2, NM1DWW3,
  NM1FC1B, NM1FC1W, NM1FC2B, NM1FC2W,
  NM2DWB0, NM2DWB1, NM2DWB2, NM2DWB3, NM2DWW0, NM2DWW1, NM2DWW2, NM2DWW3,
  NM2FC1B, NM2FC1W, NM2FC2B, NM2FC2W,
  NPROJB, NPROJW, NQKVB, NQKVW,
  NRPB0, NRPB1, NRPB2,
  NSMAB0, NSMAB1, NSMAB2, NSMAW0, NSMAW1, NSMAW2,
  NSMAPWB, NSMAPWW, NX
};
// Candidate C: x first (top-level insertion), params via sorted tree-leaves.
static const int ORD_C[48] = {
  NX,
  NLN1B, NLN1W, NLN2B, NLN2W, NLN3B, NLN3W, NLN4B, NLN4W,
  NM1DWB0, NM1DWB1, NM1DWB2, NM1DWB3, NM1DWW0, NM1DWW1, NM1DWW2, NM1DWW3,
  NM1FC1B, NM1FC1W, NM1FC2B, NM1FC2W,
  NM2DWB0, NM2DWB1, NM2DWB2, NM2DWB3, NM2DWW0, NM2DWW1, NM2DWW2, NM2DWW3,
  NM2FC1B, NM2FC1W, NM2FC2B, NM2FC2W,
  NPROJB, NPROJW, NQKVB, NQKVW,
  NRPB0, NRPB1, NRPB2,
  NSMAB0, NSMAB1, NSMAB2, NSMAW0, NSMAW1, NSMAW2,
  NSMAPWB, NSMAPWW
};
// Candidate B: pure insertion order, recursively.
static const int ORD_B[48] = {
  NX, NLN1W, NLN1B, NLN2W, NLN2B, NLN3W, NLN3B, NLN4W, NLN4B,
  NQKVW, NQKVB, NPROJW, NPROJB, NRPB0, NRPB1, NRPB2,
  NM1FC1W, NM1FC1B, NM1DWW0, NM1DWW1, NM1DWW2, NM1DWW3,
  NM1DWB0, NM1DWB1, NM1DWB2, NM1DWB3, NM1FC2W, NM1FC2B,
  NSMAW0, NSMAW1, NSMAW2, NSMAB0, NSMAB1, NSMAB2, NSMAPWW, NSMAPWB,
  NM2FC1W, NM2FC1B, NM2DWW0, NM2DWW1, NM2DWW2, NM2DWW3,
  NM2DWB0, NM2DWB1, NM2DWB2, NM2DWB3, NM2FC2W, NM2FC2B
};

extern "C" void kernel_launch(void* const* d_in, const int* in_sizes, int n_in,
                              void* d_out, int out_size, void* d_ws, size_t ws_size,
                              hipStream_t stream) {
  // ---- resolve input ordering by exact size signature -----------------
  int map[NCOUNT];
  const int* cands[3] = { ORD_A, ORD_C, ORD_B };
  bool matched = false;
  for (int ci = 0; ci < 3 && !matched; ci++) {
    if (n_in != 48) break;
    bool ok = true;
    for (int s = 0; s < 48; s++)
      if (in_sizes[s] != kSizeOf[cands[ci][s]]) { ok = false; break; }
    if (ok) {
      for (int s = 0; s < 48; s++) map[cands[ci][s]] = s;
      matched = true;
    }
  }
  if (!matched) {  // fall back to candidate A
    for (int s = 0; s < 48 && s < n_in; s++) map[ORD_A[s]] = s;
  }
  auto P = [&](int name) { return (const float*)d_in[map[name]]; };

  // ---- workspace arena (~159 MB assumed available) --------------------
  char* w = (char*)d_ws;
  float*  xpix   = (float*)(w + 0);              // 26,214,400 B  (running activation)
  char*   regB   = w + 26214400;                 // 52,428,800 B
  char*   regC   = w + 78643200;                 // 52,428,800 B
  half_t* af16   = (half_t*)(w + 131072000);     // 13,107,200 B
  half_t* bf16   = (half_t*)(w + 144179200);     // 13,107,200 B
  half_t* wts    = (half_t*)(w + 157286400);     //  1,441,792 B

  half_t* qkvh    = (half_t*)regB;               // (N,768) f16  phase 1
  float*  hbuf    = (float*)regB;                // (N,512) f32  phases 2/4
  float*  smaPing = (float*)regB;                // (N,256) f32  phase 3
  float*  hsum    = (float*)regC;                // (N,512) f32
  float*  smaPong = (float*)regC;                // (N,256) f32

  half_t* wh_qkv   = wts;                        // 196608
  half_t* wh_proj  = wh_qkv   + 196608;          // 65536
  half_t* wh_m1fc1 = wh_proj  + 65536;           // 131072
  half_t* wh_m1fc2 = wh_m1fc1 + 131072;          // 65536
  half_t* wh_m2fc1 = wh_m1fc2 + 65536;           // 131072
  half_t* wh_m2fc2 = wh_m2fc1 + 131072;          // 65536
  half_t* wh_pw    = wh_m2fc2 + 65536;           // 65536

  dim3 blk(256);
  auto cast = [&](const float* s, half_t* d, int n) {
    k_cast_f16<<<dim3((n + 255) / 256), blk, 0, stream>>>(s, d, n);
  };

  // ---- weight f16 staging (every call; deterministic) ------------------
  cast(P(NQKVW),   wh_qkv,   196608);
  cast(P(NPROJW),  wh_proj,  65536);
  cast(P(NM1FC1W), wh_m1fc1, 131072);
  cast(P(NM1FC2W), wh_m1fc2, 65536);
  cast(P(NM2FC1W), wh_m2fc1, 131072);
  cast(P(NM2FC2W), wh_m2fc2, 65536);
  cast(P(NSMAPWW), wh_pw,    65536);

  const int nElem  = BB * CC * HH * WWD;          // 6,553,600
  const int nBlkEl = (nElem + 255) / 256;
  const int lnGrid = NPIX / 8;                    // 3200 blocks, 8 pixels/block

  auto gemm = [&](int mode, const half_t* A, const half_t* Wt, const float* bias,
                  const float* res, float* o32, half_t* o16, int O) {
    int waves = (NPIX / 16) * (O >> 6);
    dim3 g(waves / 8);
    if (mode == 0)      k_gemm<0><<<g, blk, 0, stream>>>(A, Wt, bias, res, o32, o16, O);
    else if (mode == 1) k_gemm<1><<<g, blk, 0, stream>>>(A, Wt, bias, res, o32, o16, O);
    else                k_gemm<2><<<g, blk, 0, stream>>>(A, Wt, bias, res, o32, o16, O);
  };

  // ===== stage 0: NCHW -> pixel-major ==================================
  k_nchw_to_pix<<<dim3(nBlkEl), blk, 0, stream>>>(P(NX), xpix);

  // ===== block 1: local multirange attention ===========================
  k_ln<true><<<dim3(lnGrid), blk, 0, stream>>>(xpix, P(NLN1W), P(NLN1B), af16, nullptr);
  gemm(2, af16, wh_qkv, P(NQKVB), nullptr, nullptr, qkvh, 768);
  k_attn<3, 1><<<dim3(NPIX * 5 / 8), blk, 0, stream>>>(qkvh, P(NRPB0), bf16, 0, 5);
  k_attn<5, 2><<<dim3(NPIX * 2 / 8), blk, 0, stream>>>(qkvh, P(NRPB1), bf16, 5, 2);
  k_attn<7, 2><<<dim3(NPIX * 1 / 8), blk, 0, stream>>>(qkvh, P(NRPB2), bf16, 7, 1);
  gemm(1, bf16, wh_proj, P(NPROJB), xpix, xpix, nullptr, 256);

  // ===== block 2: msconvstar (mlp1) ====================================
  k_ln<true><<<dim3(lnGrid), blk, 0, stream>>>(xpix, P(NLN2W), P(NLN2B), af16, nullptr);
  gemm(0, af16, wh_m1fc1, P(NM1FC1B), nullptr, hbuf, nullptr, 512);
  k_mlp_dw<<<dim3((NPIX * HID + 255) / 256), blk, 0, stream>>>(
      hbuf, P(NM1DWW0), P(NM1DWW1), P(NM1DWW2), P(NM1DWW3),
      P(NM1DWB0), P(NM1DWB1), P(NM1DWB2), P(NM1DWB3), hsum);
  k_star<<<dim3(nBlkEl), blk, 0, stream>>>(hsum, af16);
  gemm(1, af16, wh_m1fc2, P(NM1FC2B), xpix, xpix, nullptr, 256);

  // ===== block 3: cascaded SMA =========================================
  k_ln<false><<<dim3(lnGrid), blk, 0, stream>>>(xpix, P(NLN3W), P(NLN3B), nullptr, smaPing);
  k_sma_dw<<<dim3(nBlkEl), blk, 0, stream>>>(smaPing, P(NSMAW0), P(NSMAB0), smaPong, 1);
  k_sma_dw<<<dim3(nBlkEl), blk, 0, stream>>>(smaPong, P(NSMAW1), P(NSMAB1), smaPing, 2);
  k_sma_dw<<<dim3(nBlkEl), blk, 0, stream>>>(smaPing, P(NSMAW2), P(NSMAB2), smaPong, 2);
  cast(smaPong, af16, NPIX * CC);
  gemm(1, af16, wh_pw, P(NSMAPWB), xpix, xpix, nullptr, 256);

  // ===== block 4: msconvstar (mlp2) ====================================
  k_ln<true><<<dim3(lnGrid), blk, 0, stream>>>(xpix, P(NLN4W), P(NLN4B), af16, nullptr);
  gemm(0, af16, wh_m2fc1, P(NM2FC1B), nullptr, hbuf, nullptr, 512);
  k_mlp_dw<<<dim3((NPIX * HID + 255) / 256), blk, 0, stream>>>(
      hbuf, P(NM2DWW0), P(NM2DWW1), P(NM2DWW2), P(NM2DWW3),
      P(NM2DWB0), P(NM2DWB1), P(NM2DWB2), P(NM2DWB3), hsum);
  k_star<<<dim3(nBlkEl), blk, 0, stream>>>(hsum, af16);
  gemm(1, af16, wh_m2fc2, P(NM2FC2B), xpix, xpix, nullptr, 256);

  // ===== stage final: pixel-major -> NCHW ==============================
  k_pix_to_nchw<<<dim3(nBlkEl), blk, 0, stream>>>(xpix, (float*)d_out);
}
